// RiemannFMGeodesicAttention_36258113912905
// MI455X (gfx1250) — matrix operand order repaired and checked by reading, not compiled, not run on hardware
//
#include <hip/hip_runtime.h>
#include <hip/hip_bf16.h>
#include <math.h>
#include <stdint.h>

typedef __attribute__((ext_vector_type(16))) _Float16 v16h;
typedef __attribute__((ext_vector_type(8)))  float    v8f;

#define BB   2
#define NN   512
#define DM   256
#define HH   8
#define HEADD 32
#define HALFD 16
#define XF   24
#define BN   (BB*NN)
#define EPSG 1e-6f
#define PSTR 33   // padded LDS row stride (gcd(33,64)=1 -> conflict-free)

__device__ __forceinline__ float wave_max32(float v) {
#pragma unroll
  for (int off = 16; off; off >>= 1) v = fmaxf(v, __shfl_xor(v, off, 32));
  return v;
}
__device__ __forceinline__ float wave_sum32(float v) {
#pragma unroll
  for (int off = 16; off; off >>= 1) v += __shfl_xor(v, off, 32);
  return v;
}

// -------- gfx1250 async global->LDS copy (ASYNCcnt-tracked DMA path) --------
__device__ __forceinline__ void async_b128(void* lds_ptr, const void* gptr) {
  // low 32 bits of the generic pointer to a __shared__ object == LDS offset
  unsigned lds_off = (unsigned)(uintptr_t)lds_ptr;
  unsigned long long ga = (unsigned long long)(uintptr_t)gptr;
  asm volatile("global_load_async_to_lds_b128 %0, %1, off"
               :: "v"(lds_off), "v"(ga) : "memory");
}

#if defined(__has_builtin)
#if __has_builtin(__builtin_amdgcn_s_wait_asynccnt)
#define WAIT_ASYNC(n) __builtin_amdgcn_s_wait_asynccnt(n)
#else
#define WAIT_ASYNC(n) asm volatile("s_wait_asynccnt %0" :: "n"(n) : "memory")
#endif
#else
#define WAIT_ASYNC(n) asm volatile("s_wait_asynccnt %0" :: "n"(n) : "memory")
#endif

__device__ __forceinline__ void cvt4(v16h& d, int base, float4 f) {
  d[base]     = (_Float16)f.x;
  d[base + 1] = (_Float16)f.y;
  d[base + 2] = (_Float16)f.z;
  d[base + 3] = (_Float16)f.w;
}

// ---------------------------------------------------------------------------
// Kernel 1: fused Q/K/V projection.  out = h @ W^T + b  for three weight sets.
// One wave computes one 16x16 f32 tile via v_wmma_f32_16x16x32_f16 (K=256 in
// 8 WMMA steps), operands converted f32->f16 in registers, float4 loads.
// ---------------------------------------------------------------------------
__global__ __launch_bounds__(128)
void qkv_proj_kernel(const float* __restrict__ hmat,
                     const float* __restrict__ Wq, const float* __restrict__ bq,
                     const float* __restrict__ Wk, const float* __restrict__ bk,
                     const float* __restrict__ Wv, const float* __restrict__ bv,
                     float* __restrict__ Qo, float* __restrict__ Ko,
                     float* __restrict__ Vo) {
  const int lane = threadIdx.x & 31;
  const int wv   = threadIdx.x >> 5;
  int tile = blockIdx.x * 4 + wv;           // 0..3071
  const int mat = tile >> 10;               // 0=Q 1=K 2=V
  tile &= 1023;
  const int i0 = (tile >> 4) << 4;
  const int j0 = (tile & 15) << 4;

  const float* W  = (mat == 0) ? Wq : (mat == 1) ? Wk : Wv;
  const float* bs = (mat == 0) ? bq : (mat == 1) ? bk : bv;
  float*       Op = (mat == 0) ? Qo : (mat == 1) ? Ko : Vo;

  const int half = lane >> 4;
  const int col  = lane & 15;

  v8f c = {};
  const float* arow = hmat + (size_t)(i0 + col) * DM;
  const float* brow = W    + (size_t)(j0 + col) * DM;

  for (int k0 = 0; k0 < DM; k0 += 32) {
    const float* ap = arow + k0 + 16 * half;
    const float* bp = brow + k0 + 16 * half;
    v16h a, b;
    cvt4(a, 0,  *(const float4*)(ap));
    cvt4(a, 4,  *(const float4*)(ap + 4));
    cvt4(a, 8,  *(const float4*)(ap + 8));
    cvt4(a, 12, *(const float4*)(ap + 12));
    cvt4(b, 0,  *(const float4*)(bp));
    cvt4(b, 4,  *(const float4*)(bp + 4));
    cvt4(b, 8,  *(const float4*)(bp + 8));
    cvt4(b, 12, *(const float4*)(bp + 12));
    c = __builtin_amdgcn_wmma_f32_16x16x32_f16(false, a, false, b,
                                               (short)0, c, false, false);
  }
  const float bj = bs[j0 + col];
#pragma unroll
  for (int r = 0; r < 8; ++r) {
    const int M = r + 8 * half;
    Op[(size_t)(i0 + M) * DM + j0 + col] = c[r] + bj;
  }
}

// ---------------------------------------------------------------------------
// Kernel 2: fused geometry + RoPE scores + online softmax + P@V.
// One wave per (b, head, 16-row n-tile).  m in 32-wide chunks, lane <-> m.
// V and edge_bias tiles for chunk mc+1 are double-buffered into LDS via
// global_load_async_to_lds_b128 while chunk mc's VALU work (distances,
// 16x sincos, scores) executes; s_wait_asynccnt gates consumption.
// LDS tiles use a padded row stride (33 floats) for conflict-free reads.
// P@V: one v_wmma_f32_16x16x32_f16 per chunk per 16-wide j-tile.
// ---------------------------------------------------------------------------
__global__ __launch_bounds__(32)
void attn_kernel(const float* __restrict__ Qm, const float* __restrict__ Km,
                 const float* __restrict__ Vm, const float* __restrict__ x,
                 const float* __restrict__ w_kernel,
                 const float* __restrict__ beta,
                 const float* __restrict__ edge_bias,
                 const unsigned char* __restrict__ node_mask,
                 float* __restrict__ Om) {
  __shared__ float qbuf[16 * HEADD];      // Q tile
  __shared__ float xbuf[16 * XF];         // query-node features
  __shared__ float one_m_s[16];
  __shared__ float pbuf[16 * PSTR];       // chunk probabilities (padded)
  __shared__ float row_mx[16], row_sm[16], row_rs[16];
  __shared__ float freqs[HALFD];
  __shared__ float vbuf[2][32 * PSTR];    // async-staged V chunks (padded)
  __shared__ float ebuf[2][16 * 32];      // async-staged edge_bias tiles

  const int tid = threadIdx.x;
  const int wid = blockIdx.x;             // 0..511
  const int nt  = wid & 31;
  const int hd  = (wid >> 5) & 7;
  const int b   = wid >> 8;
  const int n0  = nt << 4;
  const int half = tid >> 4;
  const int col  = tid & 15;
  const float NEG_BIG = -__builtin_inff();

  if (tid < HALFD)
    freqs[tid] = __expf(-logf(10000.f) * (float)tid / (float)HALFD);
  for (int idx = tid; idx < 16 * HEADD; idx += 32)
    qbuf[idx] = Qm[(size_t)(b * NN + n0 + (idx >> 5)) * DM + hd * HEADD + (idx & 31)];
  for (int idx = tid; idx < 16 * XF; idx += 32)
    xbuf[idx] = x[(size_t)(b * NN + n0 + idx / XF) * XF + idx % XF];
  __syncthreads();
  if (tid < 16) {
    float s = 0.f;
#pragma unroll
    for (int c2 = 0; c2 < 8; ++c2) { float v = xbuf[tid * XF + c2]; s += v * v; }
    one_m_s[tid] = 1.f - s;
    row_mx[tid] = NEG_BIG; row_sm[tid] = 0.f; row_rs[tid] = 1.f;
  }
  __syncthreads();

  const float wk0 = w_kernel[hd * 3 + 0];
  const float wk1 = w_kernel[hd * 3 + 1];
  const float wk2 = w_kernel[hd * 3 + 2];
  const float bet = beta[hd];
  const float scale = 0.17677669529663687f;   // 1/sqrt(32)

  // ---- async stage of (V chunk, edge_bias tile) for one m-chunk: 12 ops ----
  auto stage_chunk = [&](int mc, int bufid) {
    const int m0 = mc * 32;
#pragma unroll
    for (int i = 0; i < 8; ++i) {               // V: 32 rows x 32 floats
      int f = i * 128 + tid * 4;
      int row = f >> 5, cc = f & 31;            // lane's 4 floats stay in-row
      async_b128(&vbuf[bufid][row * PSTR + cc],
                 Vm + (size_t)(b * NN + m0 + row) * DM + hd * HEADD + cc);
    }
#pragma unroll
    for (int i = 0; i < 4; ++i) {               // edge_bias: 16 rows x 32 floats
      int f = i * 128 + tid * 4;
      int row = f >> 5, cc = f & 31;
      async_b128(&ebuf[bufid][f],
                 edge_bias + (size_t)(b * NN + n0 + row) * NN + m0 + cc);
    }
  };

  stage_chunk(0, 0);

  v8f acc0 = {}, acc1 = {};

  for (int mc = 0; mc < NN / 32; ++mc) {
    const int cur = mc & 1;
    const int m = mc * 32 + tid;

    if (mc + 1 < NN / 32) {
      stage_chunk(mc + 1, cur ^ 1);
      WAIT_ASYNC(12);       // async ops complete in order -> chunk mc ready
    } else {
      WAIT_ASYNC(0);
    }

    // this lane's K row (32 head dims) and geometry features
    float kreg[HEADD];
    {
      const float4* kp = (const float4*)(Km + (size_t)(b * NN + m) * DM + hd * HEADD);
#pragma unroll
      for (int q = 0; q < 8; ++q) {
        float4 f = kp[q];
        kreg[4 * q] = f.x; kreg[4 * q + 1] = f.y;
        kreg[4 * q + 2] = f.z; kreg[4 * q + 3] = f.w;
      }
    }
    if (mc + 1 < NN / 32)
      __builtin_prefetch(Km + (size_t)(b * NN + m + 32) * DM + hd * HEADD, 0, 2);

    float xm[XF];
    {
      const float* xp = x + (size_t)(b * NN + m) * XF;
#pragma unroll
      for (int c2 = 0; c2 < XF; ++c2) xm[c2] = xp[c2];
    }
    float one_m_m;
    {
      float s = 0.f;
#pragma unroll
      for (int c2 = 0; c2 < 8; ++c2) s += xm[c2] * xm[c2];
      one_m_m = 1.f - s;
    }
    const bool okm = node_mask[b * NN + m] != 0;

    for (int n = 0; n < 16; ++n) {
      const float* xn = xbuf + n * XF;
      float dh2 = 0.f, sd = 0.f, de2 = 0.f;
#pragma unroll
      for (int c2 = 0; c2 < 8; ++c2) {
        float dd = xn[c2] - xm[c2];
        dh2 += dd * dd;
        sd  += xn[8 + c2] * xm[8 + c2];
        float ee = xn[16 + c2] - xm[16 + c2];
        de2 += ee * ee;
      }
      float arg  = 1.f + 2.f * dh2 / (one_m_s[n] * one_m_m);
      float dh   = acoshf(fmaxf(arg, 1.f + EPSG));
      float sdc  = fminf(fmaxf(sd, -1.f + EPSG), 1.f - EPSG);
      float dsph = acosf(sdc);
      float dist = sqrtf(dh * dh + dsph * dsph + de2);

      float rope = 0.f;
      const float* qn = qbuf + n * HEADD;
#pragma unroll
      for (int d = 0; d < HALFD; ++d) {
        float sn, cs;
        __sincosf(dist * freqs[d], &sn, &cs);
        float q1 = qn[2 * d], q2 = qn[2 * d + 1];
        float k1 = kreg[2 * d], k2 = kreg[2 * d + 1];
        rope = fmaf(cs, q1 * k1 + q2 * k2, rope);
        rope = fmaf(sn, q1 * k2 - q2 * k1, rope);
      }
      float sc = rope * scale
               + bet * (-wk0 * dh + wk1 * sd - wk2 * de2)
               + ebuf[cur][n * 32 + tid];
      if (!okm) sc = NEG_BIG;

      float cmx   = wave_max32(sc);
      float mprev = row_mx[n];
      float mnew  = fmaxf(mprev, cmx);
      float p     = (sc < -1e37f) ? 0.f : __expf(sc - mnew);
      float psum  = wave_sum32(p);
      float rsc   = (mprev < -1e37f) ? 0.f : __expf(mprev - mnew);
      if (tid == 0) {
        row_sm[n] = row_sm[n] * rsc + psum;
        row_mx[n] = mnew;
        row_rs[n] = rsc;
      }
      pbuf[n * PSTR + tid] = p;
    }
    __syncthreads();

    // rescale accumulators by this chunk's row factors
#pragma unroll
    for (int r = 0; r < 8; ++r) {
      float rsc = row_rs[r + 8 * half];
      acc0[r] *= rsc;
      acc1[r] *= rsc;
    }

    // A = P tile (16n x 32m) from LDS (padded stride -> conflict-free)
    v16h a;
#pragma unroll
    for (int t = 0; t < 16; ++t) {
      int K = t + 16 * half;
      a[t] = (_Float16)pbuf[col * PSTR + K];
    }
    // B = V tile (32m x 16j) from async-staged LDS, two j-tiles (HEAD=32)
    v16h b0, b1;
#pragma unroll
    for (int t = 0; t < 16; ++t) {
      int K = t + 16 * half;
      const float* vrow = &vbuf[cur][K * PSTR];
      b0[t] = (_Float16)vrow[col];
      b1[t] = (_Float16)vrow[16 + col];
    }
    acc0 = __builtin_amdgcn_wmma_f32_16x16x32_f16(false, a, false, b0,
                                                  (short)0, acc0, false, false);
    acc1 = __builtin_amdgcn_wmma_f32_16x16x32_f16(false, a, false, b1,
                                                  (short)0, acc1, false, false);
    __syncthreads();
  }

#pragma unroll
  for (int r = 0; r < 8; ++r) {
    const int M = r + 8 * half;
    float sm  = row_sm[M];
    float inv = (sm > 0.f) ? 1.f / sm : 0.f;
    float* orow = Om + (size_t)(b * NN + n0 + M) * DM + hd * HEADD;
    orow[col]      = acc0[r] * inv;
    orow[16 + col] = acc1[r] * inv;
  }
}

// ---------------------------------------------------------------------------
// Kernel 3: output projection  out = O @ Wo^T + bo
// ---------------------------------------------------------------------------
__global__ __launch_bounds__(128)
void out_proj_kernel(const float* __restrict__ Om, const float* __restrict__ Wo,
                     const float* __restrict__ bo, float* __restrict__ out) {
  const int lane = threadIdx.x & 31;
  const int wv   = threadIdx.x >> 5;
  int tile = blockIdx.x * 4 + wv;           // 0..1023
  const int i0 = (tile >> 4) << 4;
  const int j0 = (tile & 15) << 4;
  const int half = lane >> 4;
  const int col  = lane & 15;

  v8f c = {};
  const float* arow = Om + (size_t)(i0 + col) * DM;
  const float* brow = Wo + (size_t)(j0 + col) * DM;

  for (int k0 = 0; k0 < DM; k0 += 32) {
    const float* ap = arow + k0 + 16 * half;
    const float* bp = brow + k0 + 16 * half;
    v16h a, b;
    cvt4(a, 0,  *(const float4*)(ap));
    cvt4(a, 4,  *(const float4*)(ap + 4));
    cvt4(a, 8,  *(const float4*)(ap + 8));
    cvt4(a, 12, *(const float4*)(ap + 12));
    cvt4(b, 0,  *(const float4*)(bp));
    cvt4(b, 4,  *(const float4*)(bp + 4));
    cvt4(b, 8,  *(const float4*)(bp + 8));
    cvt4(b, 12, *(const float4*)(bp + 12));
    c = __builtin_amdgcn_wmma_f32_16x16x32_f16(false, a, false, b,
                                               (short)0, c, false, false);
  }
  const float bj = bo[j0 + col];
#pragma unroll
  for (int r = 0; r < 8; ++r) {
    const int M = r + 8 * half;
    out[(size_t)(i0 + M) * DM + j0 + col] = c[r] + bj;
  }
}

extern "C" void kernel_launch(void* const* d_in, const int* in_sizes, int n_in,
                              void* d_out, int out_size, void* d_ws,
                              size_t ws_size, hipStream_t stream) {
  (void)in_sizes; (void)n_in; (void)out_size; (void)ws_size;
  const float* hmat = (const float*)d_in[0];
  const float* x    = (const float*)d_in[1];
  const float* Wq = (const float*)d_in[2];  const float* bq = (const float*)d_in[3];
  const float* Wk = (const float*)d_in[4];  const float* bk = (const float*)d_in[5];
  const float* Wv = (const float*)d_in[6];  const float* bv = (const float*)d_in[7];
  const float* Wo = (const float*)d_in[8];  const float* bo = (const float*)d_in[9];
  const float* wker = (const float*)d_in[10];
  const float* beta = (const float*)d_in[11];
  const float* eb   = (const float*)d_in[12];
  const unsigned char* mask = (const unsigned char*)d_in[13];

  float* ws = (float*)d_ws;
  float* Q = ws;
  float* K = ws + (size_t)1 * BN * DM;
  float* V = ws + (size_t)2 * BN * DM;
  float* O = ws + (size_t)3 * BN * DM;

  hipLaunchKernelGGL(qkv_proj_kernel, dim3(768), dim3(128), 0, stream,
                     hmat, Wq, bq, Wk, bk, Wv, bv, Q, K, V);
  hipLaunchKernelGGL(attn_kernel, dim3(512), dim3(32), 0, stream,
                     Q, K, V, x, wker, beta, eb, mask, O);
  hipLaunchKernelGGL(out_proj_kernel, dim3(256), dim3(128), 0, stream,
                     O, Wo, bo, (float*)d_out);
}
